// MultiPositiveSimCLR_35055523070221
// MI455X (gfx1250) — matrix operand order repaired and checked by reading, not compiled, not run on hardware
//
#include <hip/hip_runtime.h>
#include <hip/hip_bf16.h>
#include <math.h>

typedef __attribute__((ext_vector_type(16))) _Float16 v16h;
typedef __attribute__((ext_vector_type(8)))  _Float16 v8h;
typedef __attribute__((ext_vector_type(8)))  float    v8f;

#define B_       2048
#define NPOS_    3
#define D_       128
#define N_       8192               // B_*(1+NPOS_)
#define XROWS_   (NPOS_ * B_)       // 6144
#define INV_TEMP 10.0f
#define NNEG_    4096
// INV_TEMP * log2(e): exp(sim - 10) == exp2(fma(cos, K2, -K2))
#define K2_      14.4269504088896340736f
#define LN2_     0.69314718055994530942f
// ln(NNEG / (N-1)) : expectation-equivalent correction for random subsample
#define SUBSAMPLE_LOG_ -0.6930251f

// ---------------------------------------------------------------------------
// Kernel 1: L2-normalize rows of z = [x ; x_pair] -> f16 zn[N_][D_]
// One wave32 per row; each lane owns 4 contiguous floats.
// ---------------------------------------------------------------------------
__global__ __launch_bounds__(256) void k_normalize(const float* __restrict__ x,
                                                   const float* __restrict__ xp,
                                                   _Float16* __restrict__ zn) {
    const int wave = threadIdx.x >> 5;
    const int lane = threadIdx.x & 31;
    const int row  = blockIdx.x * 8 + wave;

    const float* src = (row < XROWS_) ? (x + (size_t)row * D_)
                                      : (xp + (size_t)(row - XROWS_) * D_);
    float4 v = reinterpret_cast<const float4*>(src)[lane];
    float ss = v.x * v.x + v.y * v.y + v.z * v.z + v.w * v.w;
#pragma unroll
    for (int m = 16; m >= 1; m >>= 1) ss += __shfl_xor(ss, m, 32);
    float r = rsqrtf(ss);

    union { _Float16 h[4]; uint2 u; } o;
    o.h[0] = (_Float16)(v.x * r);
    o.h[1] = (_Float16)(v.y * r);
    o.h[2] = (_Float16)(v.z * r);
    o.h[3] = (_Float16)(v.w * r);
    reinterpret_cast<uint2*>(zn + (size_t)row * D_)[lane] = o.u;
}

// ---------------------------------------------------------------------------
// Kernel 2: fused sim = (zn @ zn^T)/TEMP + row sum-of-exp with FIXED shift 10
// (cosines are bounded by 1 -> sim <= 10, so no online max is needed).
// Grid: 512 blocks (16-row tiles) x 256 threads (8 wave32).
// Wave w sweeps columns [w*1024,(w+1)*1024) in 16-wide tiles, 4 WMMA per tile,
// with register double-buffered B fragments (loads one tile ahead).
// Branchless per-element: 1 fma + 1 v_exp_f32 + 1 cndmask (diag) + 1 add.
// ---------------------------------------------------------------------------
__global__ __launch_bounds__(256) void k_simlse(const _Float16* __restrict__ zn,
                                                float* __restrict__ row_lse) {
    __shared__ float lds_s[8][16];

    const int wave = threadIdx.x >> 5;
    const int lane = threadIdx.x & 31;
    const int h    = lane >> 4;     // half-wave select
    const int mloc = lane & 15;     // A-row index / C-column index
    const int i0   = blockIdx.x * 16;
    const int row0 = i0 + 8 * h;    // first of the 8 C-rows this lane holds

    // A fragments for K = 0..127 (4 x 32), loaded once, reused for 64 tiles.
    // A k order used here (same as B): lane (h,m): elems j = K 16h+j -> contiguous 32B
    v16h a[4];
    {
        const _Float16* arow = zn + (size_t)(i0 + mloc) * D_;
#pragma unroll
        for (int kt = 0; kt < 4; ++kt) {
            a[kt] = *reinterpret_cast<const v16h*>(arow + kt * 32 + 16 * h);
        }
    }

    float rs[8];
#pragma unroll
    for (int r = 0; r < 8; ++r) rs[r] = 0.0f;

    const int jbase = wave * (N_ / 8);
    // B k order used here (same as A): lane (h,n): elems j = K 16h+j -> contiguous 32B
    const _Float16* bbase = zn + (size_t)(jbase + mloc) * D_ + 16 * h;

    v16h b0[4], b1[4];
#pragma unroll
    for (int kt = 0; kt < 4; ++kt)
        b0[kt] = *reinterpret_cast<const v16h*>(bbase + kt * 32);

    for (int t = 0; t < 64; t += 2) {
        // Issue loads for tile t+1 before consuming tile t.
        const _Float16* pn1 = bbase + (size_t)(t + 1) * 16 * D_;
#pragma unroll
        for (int kt = 0; kt < 4; ++kt)
            b1[kt] = *reinterpret_cast<const v16h*>(pn1 + kt * 32);
        __builtin_prefetch(pn1 + 32 * D_, 0, 0);   // global_prefetch_b8, tile t+3

        // ---- tile t ----
        {
            v8f c = {};
#pragma unroll
            for (int kt = 0; kt < 4; ++kt)
                c = __builtin_amdgcn_wmma_f32_16x16x32_f16(
                        false, a[kt], false, b0[kt], (short)0, c, false, false);
            const int dr = (jbase + t * 16 + mloc) - row0;  // diag row or OOR
#pragma unroll
            for (int r = 0; r < 8; ++r) {
                const float e = __builtin_amdgcn_exp2f(fmaf(c[r], K2_, -K2_));
                rs[r] += (r == dr) ? 0.0f : e;
            }
        }

        // Issue loads for tile t+2 before consuming tile t+1.
        if (t + 2 < 64) {
            const _Float16* pn2 = bbase + (size_t)(t + 2) * 16 * D_;
#pragma unroll
            for (int kt = 0; kt < 4; ++kt)
                b0[kt] = *reinterpret_cast<const v16h*>(pn2 + kt * 32);
        }

        // ---- tile t+1 ----
        {
            v8f c = {};
#pragma unroll
            for (int kt = 0; kt < 4; ++kt)
                c = __builtin_amdgcn_wmma_f32_16x16x32_f16(
                        false, a[kt], false, b1[kt], (short)0, c, false, false);
            const int dr = (jbase + (t + 1) * 16 + mloc) - row0;
#pragma unroll
            for (int r = 0; r < 8; ++r) {
                const float e = __builtin_amdgcn_exp2f(fmaf(c[r], K2_, -K2_));
                rs[r] += (r == dr) ? 0.0f : e;
            }
        }
    }

    // Sum across the 16 lanes of each half (same rows, different columns).
#pragma unroll
    for (int msk = 1; msk < 16; msk <<= 1)
#pragma unroll
        for (int r = 0; r < 8; ++r) rs[r] += __shfl_xor(rs[r], msk, 32);

    if (mloc == 0)
#pragma unroll
        for (int r = 0; r < 8; ++r) lds_s[wave][8 * h + r] = rs[r];
    __syncthreads();

    // One thread per row sums the 8 column-split partials; lse = 10 + ln(S).
    if (threadIdx.x < 16) {
        float S = 0.0f;
#pragma unroll
        for (int w = 0; w < 8; ++w) S += lds_s[w][threadIdx.x];
        row_lse[i0 + threadIdx.x] =
            fmaf(__builtin_amdgcn_logf(S), LN2_, INV_TEMP);
    }
}

// ---------------------------------------------------------------------------
// Kernel 3: final scalar. loss_neg = mean(row_lse) + ln(NNEG/(N-1)),
// loss_pos = WEIGHT * sum(dot(xn_pb, pn_b)) / TEMP / (B*NPOS).
// Single block of 256 threads; no atomics.
// ---------------------------------------------------------------------------
__global__ __launch_bounds__(256) void k_finalize(const _Float16* __restrict__ zn,
                                                  const float* __restrict__ row_lse,
                                                  float* __restrict__ out) {
    __shared__ float sred[256];
    const int t = threadIdx.x;

    float lsum = 0.0f;
    for (int i = t; i < N_; i += 256) lsum += row_lse[i];

    float psum = 0.0f;
    for (int pair = t; pair < XROWS_; pair += 256) {
        const int b = pair & (B_ - 1);
        const _Float16* pa = zn + (size_t)pair * D_;
        const _Float16* pb = zn + (size_t)(XROWS_ + b) * D_;
        float d = 0.0f;
#pragma unroll 8
        for (int k = 0; k < D_; ++k) d += (float)pa[k] * (float)pb[k];
        psum += d;
    }

    sred[t] = lsum;
    __syncthreads();
    for (int s = 128; s; s >>= 1) { if (t < s) sred[t] += sred[t + s]; __syncthreads(); }
    float lossNeg = 0.0f;
    if (t == 0) lossNeg = sred[0] / (float)N_ + SUBSAMPLE_LOG_;
    __syncthreads();

    sred[t] = psum;
    __syncthreads();
    for (int s = 128; s; s >>= 1) { if (t < s) sred[t] += sred[t + s]; __syncthreads(); }
    if (t == 0) {
        const float lossPos = (sred[0] * INV_TEMP) / (float)(B_ * NPOS_); // WEIGHT=1
        out[0] = lossNeg - lossPos;
    }
}

// ---------------------------------------------------------------------------
extern "C" void kernel_launch(void* const* d_in, const int* in_sizes, int n_in,
                              void* d_out, int out_size, void* d_ws, size_t ws_size,
                              hipStream_t stream) {
    (void)in_sizes; (void)n_in; (void)out_size; (void)ws_size;
    const float* x  = (const float*)d_in[0];   // [NPOS*B, D] f32
    const float* xp = (const float*)d_in[1];   // [B, D] f32
    float* out = (float*)d_out;

    // Workspace layout: zn f16 [N_, D_] (2 MB) | row_lse f32 [N_] (32 KB)
    _Float16* zn      = (_Float16*)d_ws;
    float*    row_lse = (float*)((char*)d_ws + (size_t)N_ * D_ * sizeof(_Float16));

    k_normalize<<<N_ / 8, 256, 0, stream>>>(x, xp, zn);
    k_simlse  <<<N_ / 16, 256, 0, stream>>>(zn, row_lse);
    k_finalize<<<1, 256, 0, stream>>>(zn, row_lse, out);
}
